// RelMHAtt_44753559225088
// MI455X (gfx1250) — matrix operand order repaired and checked by reading, not compiled, not run on hardware
//
#include <hip/hip_runtime.h>
#include <stdint.h>

// ---------------------------------------------------------------------------
// RelMHAtt for MI455X (gfx1250): fused f16-WMMA pipeline.
//   dims: B=4, S=512, HID=1024, H=16 heads, D=64, REL=64
// ---------------------------------------------------------------------------

typedef __attribute__((ext_vector_type(16))) _Float16 v16h;
typedef __attribute__((ext_vector_type(8)))  _Float16 v8h;
typedef __attribute__((ext_vector_type(8)))  float    v8f;
typedef __attribute__((ext_vector_type(4)))  float    vf4;

#define DEVINL __device__ __forceinline__

static constexpr int Bn   = 4;
static constexpr int S    = 512;
static constexpr int H    = 16;
static constexpr int D    = 64;
static constexpr int HID  = 1024;
static constexpr int NTOK = Bn * S;      // 2048
static constexpr int REL  = 64;

// ---------------------------------------------------------------------------
// helpers
// ---------------------------------------------------------------------------
DEVINL v8f zero8() {
  v8f z;
#pragma unroll
  for (int i = 0; i < 8; ++i) z[i] = 0.0f;
  return z;
}

DEVINL v8f wmma_f16(v16h a, v16h b, v8f c) {
  // D(16x16,f32) = A(16x32,f16) * B(32x16,f16) + C
  return __builtin_amdgcn_wmma_f32_16x16x32_f16(false, a, false, b,
                                                (short)0, c, false, false);
}

// per-wave LDS staging sync: DS ops are in-order within a wave on CDNA5;
// we only need a compiler scheduling barrier + a dscnt drain.
DEVINL void wave_sync_lds() {
  __builtin_amdgcn_wave_barrier();
  asm volatile("s_wait_dscnt 0" ::: "memory");
  __builtin_amdgcn_wave_barrier();
}

// A-fragment (16x32 f16): lane l holds row (l&15); elems 0..7 = K (l>>4)*8+0..7,
// elems 8..15 = K 16+(l>>4)*8+0..7.  Source: row-major f16, leading dim ld.
DEVINL v16h load_a16(const _Float16* tile, int ld, int kbase, int lane) {
  const int half = lane >> 4, r = lane & 15;
  const _Float16* p = tile + (long)r * ld + kbase + half * 8;
  v8h lo = *(const v8h*)p;
  v8h hi = *(const v8h*)(p + 16);
  v16h a;
#pragma unroll
  for (int i = 0; i < 8; ++i) { a[i] = lo[i]; a[8 + i] = hi[i]; }
  return a;
}

// Same A pattern, but converting from f32 source (used for rel_embed).
DEVINL v16h load_a32cvt(const float* tile, int ld, int kbase, int lane) {
  const int half = lane >> 4, r = lane & 15;
  const float* p = tile + (long)r * ld + kbase + half * 8;
  vf4 x0 = *(const vf4*)p;
  vf4 x1 = *(const vf4*)(p + 4);
  vf4 y0 = *(const vf4*)(p + 16);
  vf4 y1 = *(const vf4*)(p + 20);
  v16h a;
#pragma unroll
  for (int i = 0; i < 4; ++i) {
    a[i]      = (_Float16)x0[i];
    a[4 + i]  = (_Float16)x1[i];
    a[8 + i]  = (_Float16)y0[i];
    a[12 + i] = (_Float16)y1[i];
  }
  return a;
}

// B-fragment (32x16 f16): lane l holds column N (l&15); elems 0..7 = K (l>>4)*8+0..7,
// elems 8..15 = K 16+(l>>4)*8+0..7.  Source must be [K][N] row-major f16, leading dim ld.
DEVINL v16h load_b16(const _Float16* base, int ld, int k0, int n0, int lane) {
  const int half = lane >> 4, n = lane & 15;
  const _Float16* p = base + (long)(k0 + half * 8) * ld + n0 + n;
  v16h b;
#pragma unroll
  for (int i = 0; i < 8; ++i) { b[i] = p[(long)i * ld]; b[8 + i] = p[(long)(16 + i) * ld]; }
  return b;
}

// ---------------------------------------------------------------------------
// kernel 0a: f32 -> f16 copy
// ---------------------------------------------------------------------------
__global__ void cvt_f16_kernel(const float* __restrict__ src,
                               _Float16* __restrict__ dst, long n) {
  long i = (long)blockIdx.x * blockDim.x + threadIdx.x;
  long stride = (long)gridDim.x * blockDim.x;
  for (; i < n; i += stride) dst[i] = (_Float16)src[i];
}

// kernel 0b: f32 [R][C] -> f16 [C][R] transposing convert
__global__ void cvt_tr_kernel(const float* __restrict__ src,
                              _Float16* __restrict__ dst, int R, int C) {
  long n = (long)R * C;
  long i = (long)blockIdx.x * blockDim.x + threadIdx.x;
  long stride = (long)gridDim.x * blockDim.x;
  for (; i < n; i += stride) {
    long r = i / C, c = i % C;
    dst[c * (long)R + r] = (_Float16)src[i];
  }
}

// ---------------------------------------------------------------------------
// kernel 1: projections  Y = X @ W^T + b  for q/k/v  (NT GEMM, f16 WMMA)
//   q -> qh  [b,h,s,d]    (row-major, A-side of QK^T and store target)
//   k -> khT [b,h,d,s]    (transposed: B-side of QK^T)
//   v -> vh  [b,h,s,d]    (row-major: B-side of P*V)
// one wave computes a 64x64 output tile; 8 waves / block.
// ---------------------------------------------------------------------------
__global__ __launch_bounds__(256) void proj_kernel(
    const _Float16* __restrict__ Xq, const _Float16* __restrict__ Xk,
    const _Float16* __restrict__ Xv, const _Float16* __restrict__ WqT,
    const _Float16* __restrict__ WkT, const _Float16* __restrict__ WvT,
    const float* __restrict__ bq, const float* __restrict__ bk,
    const float* __restrict__ bv, _Float16* __restrict__ qh,
    _Float16* __restrict__ khT, _Float16* __restrict__ vh) {
  __shared__ _Float16 stage[8][16 * 16];
  const int wave = threadIdx.x >> 5, lane = threadIdx.x & 31;
  const int tile = blockIdx.x * 8 + wave;          // 0..1535
  const int p  = tile / 512;                       // 0=q 1=k 2=v
  const int t  = tile % 512;
  const int mt = t >> 4, nt = t & 15;              // 32 x 16 tiles
  const int m0 = mt * 64, n0 = nt * 64;
  const int head = nt;                             // 64-wide N tile == one head
  const int bidx = m0 / S, s_base = m0 % S;

  const _Float16* X  = (p == 0) ? Xq : (p == 1) ? Xk : Xv;
  const _Float16* WT = (p == 0) ? WqT : (p == 1) ? WkT : WvT;
  const float*    bs = (p == 0) ? bq : (p == 1) ? bk : bv;

  v8f c[4][4];
#pragma unroll
  for (int i = 0; i < 4; ++i)
#pragma unroll
    for (int j = 0; j < 4; ++j) c[i][j] = zero8();

  for (int ks = 0; ks < HID; ks += 32) {
    v16h a[4];
#pragma unroll
    for (int mi = 0; mi < 4; ++mi)
      a[mi] = load_a16(X + (long)(m0 + mi * 16) * HID, HID, ks, lane);
#pragma unroll
    for (int ni = 0; ni < 4; ++ni) {
      v16h bfr = load_b16(WT, HID, ks, n0 + ni * 16, lane);
#pragma unroll
      for (int mi = 0; mi < 4; ++mi) c[mi][ni] = wmma_f16(a[mi], bfr, c[mi][ni]);
    }
  }

  const int n_lane = lane & 15, halfl = lane >> 4;
  _Float16* st = stage[wave];
#pragma unroll
  for (int mi = 0; mi < 4; ++mi) {
#pragma unroll
    for (int ni = 0; ni < 4; ++ni) {
      const float bb = bs[n0 + ni * 16 + n_lane];
      v8f acc = c[mi][ni];
#pragma unroll
      for (int i = 0; i < 8; ++i) {
        const int m = i + 8 * halfl;               // D-frag row
        st[m * 16 + n_lane] = (_Float16)(acc[i] + bb);
      }
      wave_sync_lds();
      if (p != 1) {                                // row-major [b,h,s,d]
        const int row = lane >> 1, c0 = (lane & 1) * 8;
        v8h vvec;
#pragma unroll
        for (int i = 0; i < 8; ++i) vvec[i] = st[row * 16 + c0 + i];
        _Float16* dst = ((p == 0) ? qh : vh) +
            ((long)(bidx * H + head) * S + (s_base + mi * 16 + row)) * D +
            ni * 16 + c0;
        *(v8h*)dst = vvec;
      } else {                                     // transposed [b,h,d,s]
        const int col = lane >> 1, r0 = (lane & 1) * 8;
        v8h vvec;
#pragma unroll
        for (int i = 0; i < 8; ++i) vvec[i] = st[(r0 + i) * 16 + col];
        _Float16* dst = khT +
            ((long)(bidx * H + head) * D + ni * 16 + col) * S +
            (s_base + mi * 16 + r0);
        *(v8h*)dst = vvec;
      }
      wave_sync_lds();
    }
  }
}

// ---------------------------------------------------------------------------
// kernel 2: relative-position bias
//   rbias[b,h,q,k] = log(max(relu(rel[b,q,k,:] @ Wr^T + br), 1e-6))  (f16)
// One wave: 16 consecutive k-rows x 16 heads, K=REL=64 (2 WMMAs).
// Reads rel_embed (256 MB) exactly once; output stores are one 16B/lane.
// ---------------------------------------------------------------------------
__global__ __launch_bounds__(256) void relbias_kernel(
    const float* __restrict__ rel, const _Float16* __restrict__ WrT,
    const float* __restrict__ br, _Float16* __restrict__ rbias) {
  const int wave = threadIdx.x >> 5, lane = threadIdx.x & 31;
  const long wid = (long)blockIdx.x * 8 + wave;    // 0..65535
  const int kt = (int)(wid & 31);
  const int q  = (int)((wid >> 5) & 511);
  const int b  = (int)(wid >> 14);

  const float* A = rel + (((long)(b * S + q) * S) + kt * 16) * REL;
  v16h b0 = load_b16(WrT, H, 0, 0, lane);
  v16h b1 = load_b16(WrT, H, 32, 0, lane);
  v16h a0 = load_a32cvt(A, REL, 0, lane);
  v16h a1 = load_a32cvt(A, REL, 32, lane);
  v8f c = zero8();
  c = wmma_f16(a0, b0, c);
  c = wmma_f16(a1, b1, c);

  const int hh = lane & 15, half = lane >> 4;
  const float bb = br[hh];
  v8h o;
#pragma unroll
  for (int i = 0; i < 8; ++i) {
    float v = c[i] + bb;
    v = fmaxf(v, 1e-6f);                           // relu then clip(min=1e-6)
    o[i] = (_Float16)__logf(v);
  }
  _Float16* dst = rbias + (((long)(b * H + hh) * S + q) * S) + kt * 16 + half * 8;
  *(v8h*)dst = o;
}

// ---------------------------------------------------------------------------
// kernel 3: attention core.  4 waves/block; each wave owns 16 q rows of one
// (b,h).  LDS per wave: scoreT[512][16] f32, P[16][512] f16, stats, staging.
// ---------------------------------------------------------------------------
static constexpr int SCORE_BYTES = S * 16 * 4;        // 32768
static constexpr int PBUF_BYTES  = 16 * S * 2;        // 16384
static constexpr int STATS_BYTES = 64;
static constexpr int STG_BYTES   = 16 * 16 * 2;       // 512
static constexpr int WAVE_LDS    = SCORE_BYTES + PBUF_BYTES + STATS_BYTES + STG_BYTES;
static constexpr int ATTN_LDS    = WAVE_LDS * 4;      // 198912 (< 320KB WGP)

__global__ __launch_bounds__(128) void attn_kernel(
    const _Float16* __restrict__ qh, const _Float16* __restrict__ khT,
    const _Float16* __restrict__ vh, const _Float16* __restrict__ rbias,
    const unsigned char* __restrict__ mask, _Float16* __restrict__ att) {
  extern __shared__ char smem[];
  const int wave = threadIdx.x >> 5, lane = threadIdx.x & 31;
  const int bid = blockIdx.x;                       // 512 blocks
  const int qt = bid & 7, h = (bid >> 3) & 15, b = bid >> 7;
  const int bh = b * H + h;
  const int q0 = qt * 64 + wave * 16;

  float*    scoreT = (float*)(smem + wave * WAVE_LDS);
  _Float16* P      = (_Float16*)(smem + wave * WAVE_LDS + SCORE_BYTES);
  float*    stats  = (float*)(smem + wave * WAVE_LDS + SCORE_BYTES + PBUF_BYTES);
  _Float16* stg    = (_Float16*)(smem + wave * WAVE_LDS + SCORE_BYTES +
                                 PBUF_BYTES + STATS_BYTES);

  // ---- phase A: raw scores = (Q K^T) / 8, written transposed to LDS ----
  const _Float16* Q  = qh + ((long)bh * S + q0) * D;
  const _Float16* KT = khT + (long)bh * D * S;
  v16h aq0 = load_a16(Q, D, 0, lane);
  v16h aq1 = load_a16(Q, D, 32, lane);
  const int nl = lane & 15, half = lane >> 4;
  for (int kt = 0; kt < S / 16; ++kt) {
    v16h bk0 = load_b16(KT, S, 0, kt * 16, lane);
    v16h bk1 = load_b16(KT, S, 32, kt * 16, lane);
    v8f c = zero8();
    c = wmma_f16(aq0, bk0, c);
    c = wmma_f16(aq1, bk1, c);
    float* d = scoreT + (kt * 16 + nl) * 16 + half * 8;   // [k][m], m contig
    vf4 lo, hi;
#pragma unroll
    for (int i = 0; i < 4; ++i) { lo[i] = c[i] * 0.125f; hi[i] = c[4 + i] * 0.125f; }
    *(vf4*)d = lo;
    *(vf4*)(d + 4) = hi;
  }
  wave_sync_lds();

  // ---- phase B: softmax (exact, two-pass).  2 lanes per q-row. ----
  const int row = lane & 15;                        // q-row within tile
  const int kh0 = half * 256;                       // this lane's k range
  const _Float16* rb = rbias + ((long)bh * S + q0 + row) * S + kh0;
  const unsigned char* mk = mask + b * S + kh0;
  float mrow = -1e30f;
  for (int j0 = 0; j0 < 256; j0 += 8) {
    v8h r8 = *(const v8h*)(rb + j0);
#pragma unroll
    for (int e = 0; e < 8; ++e) {
      const int k = kh0 + j0 + e;
      float v = scoreT[k * 16 + row] + (float)r8[e];
      if (mk[j0 + e]) v = -1e9f;
      scoreT[k * 16 + row] = v;                     // keep combined score
      mrow = fmaxf(mrow, v);
    }
  }
  mrow = fmaxf(mrow, __shfl_xor(mrow, 16, 32));
  float sum = 0.0f;
  for (int j0 = 0; j0 < 256; j0 += 8) {
    v8h p8;
#pragma unroll
    for (int e = 0; e < 8; ++e) {
      const float v = scoreT[(kh0 + j0 + e) * 16 + row];
      const float pe = __expf(v - mrow);
      sum += pe;
      p8[e] = (_Float16)pe;
    }
    *(v8h*)(P + (long)row * S + kh0 + j0) = p8;     // P row-major f16
  }
  sum += __shfl_xor(sum, 16, 32);
  if (lane < 16) stats[lane] = 1.0f / sum;          // lanes 0..15: row==lane
  wave_sync_lds();

  // ---- phase C: out = P @ V (unnormalized), rescale by 1/sum at the end ----
  const _Float16* V = vh + (long)bh * S * D;
  v8f o[4];
#pragma unroll
  for (int i = 0; i < 4; ++i) o[i] = zero8();
  for (int kt = 0; kt < S / 32; ++kt) {
    v16h ap = load_a16(P, S, kt * 32, lane);        // A from LDS
#pragma unroll
    for (int ni = 0; ni < 4; ++ni) {
      v16h bv_ = load_b16(V, D, kt * 32, ni * 16, lane);
      o[ni] = wmma_f16(ap, bv_, o[ni]);
    }
  }
  float inv[8];
#pragma unroll
  for (int i = 0; i < 8; ++i) inv[i] = stats[i + 8 * half];
#pragma unroll
  for (int ni = 0; ni < 4; ++ni) {
#pragma unroll
    for (int i = 0; i < 8; ++i)
      stg[(i + 8 * half) * 16 + nl] = (_Float16)(o[ni][i] * inv[i]);
    wave_sync_lds();
    const int rrow = lane >> 1, c0 = (lane & 1) * 8;
    v8h vvec;
#pragma unroll
    for (int i = 0; i < 8; ++i) vvec[i] = stg[rrow * 16 + c0 + i];
    _Float16* dst = att + (long)(b * S + q0 + rrow) * HID + h * D + ni * 16 + c0;
    *(v8h*)dst = vvec;
    wave_sync_lds();
  }
}

// ---------------------------------------------------------------------------
// kernel 4: output projection  out = att @ Wm^T + bm   (f32 output)
// ---------------------------------------------------------------------------
__global__ __launch_bounds__(256) void outproj_kernel(
    const _Float16* __restrict__ A16, const _Float16* __restrict__ WmT,
    const float* __restrict__ bm, float* __restrict__ out) {
  const int wave = threadIdx.x >> 5, lane = threadIdx.x & 31;
  const int tile = blockIdx.x * 8 + wave;           // 0..511
  const int mt = tile >> 4, nt = tile & 15;
  const int m0 = mt * 64, n0 = nt * 64;

  v8f c[4][4];
#pragma unroll
  for (int i = 0; i < 4; ++i)
#pragma unroll
    for (int j = 0; j < 4; ++j) c[i][j] = zero8();

  for (int ks = 0; ks < HID; ks += 32) {
    v16h a[4];
#pragma unroll
    for (int mi = 0; mi < 4; ++mi)
      a[mi] = load_a16(A16 + (long)(m0 + mi * 16) * HID, HID, ks, lane);
#pragma unroll
    for (int ni = 0; ni < 4; ++ni) {
      v16h bfr = load_b16(WmT, HID, ks, n0 + ni * 16, lane);
#pragma unroll
      for (int mi = 0; mi < 4; ++mi) c[mi][ni] = wmma_f16(a[mi], bfr, c[mi][ni]);
    }
  }
  const int nl = lane & 15, half = lane >> 4;
#pragma unroll
  for (int mi = 0; mi < 4; ++mi)
#pragma unroll
    for (int ni = 0; ni < 4; ++ni) {
      const float bb = bm[n0 + ni * 16 + nl];
#pragma unroll
      for (int i = 0; i < 8; ++i) {
        const long m = m0 + mi * 16 + i + 8 * half;
        out[m * HID + n0 + ni * 16 + nl] = c[mi][ni][i] + bb;
      }
    }
}

// ---------------------------------------------------------------------------
// launcher
// ---------------------------------------------------------------------------
extern "C" void kernel_launch(void* const* d_in, const int* in_sizes, int n_in,
                              void* d_out, int out_size, void* d_ws,
                              size_t ws_size, hipStream_t stream) {
  (void)in_sizes; (void)n_in; (void)out_size; (void)ws_size;
  const float* v_in  = (const float*)d_in[0];
  const float* k_in  = (const float*)d_in[1];
  const float* q_in  = (const float*)d_in[2];
  const unsigned char* mask = (const unsigned char*)d_in[3];
  const float* rel   = (const float*)d_in[4];
  const float* Wv = (const float*)d_in[5];  const float* bv = (const float*)d_in[6];
  const float* Wk = (const float*)d_in[7];  const float* bk = (const float*)d_in[8];
  const float* Wq = (const float*)d_in[9];  const float* bq = (const float*)d_in[10];
  const float* Wr = (const float*)d_in[11]; const float* br = (const float*)d_in[12];
  const float* Wm = (const float*)d_in[13]; const float* bm = (const float*)d_in[14];
  float* out = (float*)d_out;

  char* w = (char*)d_ws;
  size_t off = 0;
  auto alloc = [&](size_t bytes) {
    void* p = w + off;
    off = (off + bytes + 255) & ~(size_t)255;
    return p;
  };
  const size_t XB = (size_t)NTOK * HID * 2;   // 4 MB
  const size_t WB = (size_t)HID * HID * 2;    // 2 MB
  _Float16* Xq16  = (_Float16*)alloc(XB);
  _Float16* Xk16  = (_Float16*)alloc(XB);
  _Float16* Xv16  = (_Float16*)alloc(XB);
  _Float16* WqT16 = (_Float16*)alloc(WB);
  _Float16* WkT16 = (_Float16*)alloc(WB);
  _Float16* WvT16 = (_Float16*)alloc(WB);
  _Float16* WmT16 = (_Float16*)alloc(WB);
  _Float16* WrT16 = (_Float16*)alloc((size_t)REL * H * 2);
  _Float16* qh    = (_Float16*)alloc((size_t)Bn * H * S * D * 2);  // 4 MB
  _Float16* khT   = (_Float16*)alloc((size_t)Bn * H * D * S * 2);  // 4 MB
  _Float16* vh    = (_Float16*)alloc((size_t)Bn * H * S * D * 2);  // 4 MB
  _Float16* rbias = (_Float16*)alloc((size_t)Bn * H * S * S * 2);  // 32 MB
  _Float16* att16 = (_Float16*)alloc(XB);                          // 4 MB

  const long NX = (long)NTOK * HID;
  cvt_f16_kernel<<<1024, 256, 0, stream>>>(v_in, Xv16, NX);
  cvt_f16_kernel<<<1024, 256, 0, stream>>>(k_in, Xk16, NX);
  cvt_f16_kernel<<<1024, 256, 0, stream>>>(q_in, Xq16, NX);
  cvt_tr_kernel<<<1024, 256, 0, stream>>>(Wq, WqT16, HID, HID);
  cvt_tr_kernel<<<1024, 256, 0, stream>>>(Wk, WkT16, HID, HID);
  cvt_tr_kernel<<<1024, 256, 0, stream>>>(Wv, WvT16, HID, HID);
  cvt_tr_kernel<<<1024, 256, 0, stream>>>(Wm, WmT16, HID, HID);
  cvt_tr_kernel<<<4, 256, 0, stream>>>(Wr, WrT16, H, REL);

  proj_kernel<<<192, 256, 0, stream>>>(Xq16, Xk16, Xv16, WqT16, WkT16, WvT16,
                                       bq, bk, bv, qh, khT, vh);
  relbias_kernel<<<8192, 256, 0, stream>>>(rel, WrT16, br, rbias);
  attn_kernel<<<512, 128, ATTN_LDS, stream>>>(qh, khT, vh, rbias, mask, att16);
  outproj_kernel<<<64, 256, 0, stream>>>(att16, WmT16, bm, out);
}
